// MoEPositionwiseFFN_34136400068821
// MI455X (gfx1250) — matrix operand order repaired and checked by reading, not compiled, not run on hardware
//
#include <hip/hip_runtime.h>

#define S   4096      // B*T tokens
#define Dm  1024      // model dim
#define Fm  4096      // ffn dim
#define Em  8         // experts
#define TM  128       // GEMM row tile
#define CAP 9216      // S*K + Em*TM padded assignment capacity
#define LDT 40        // LDS tile row stride (elements), 80B -> 16B aligned chunks
#define BUFB (TM * LDT * 2)   // bytes per LDS tile buffer (10240)

typedef __attribute__((ext_vector_type(16))) __bf16 v16bf;
typedef __attribute__((ext_vector_type(8)))  __bf16 v8bf;
typedef __attribute__((ext_vector_type(8)))  float  v8f;

union ABFrag { v16bf v; v8bf h[2]; };

__device__ __forceinline__ unsigned short f32_to_bf16(float f) {
  union { float f; unsigned int u; } c; c.f = f;
  unsigned int r = c.u + 0x7FFFu + ((c.u >> 16) & 1u);   // round-to-nearest-even
  return (unsigned short)(r >> 16);
}

// 16B async copy global -> LDS (ASYNCcnt-tracked, no VGPR staging)
__device__ __forceinline__ void async_ld16(unsigned lds_addr, const void* gaddr) {
  asm volatile("global_load_async_to_lds_b128 %0, %1, off"
               :: "v"(lds_addr), "v"((unsigned long long)(uintptr_t)gaddr)
               : "memory");
}
__device__ __forceinline__ void wait_async_le2() {
  asm volatile("s_wait_asynccnt 0x2" ::: "memory");
}
__device__ __forceinline__ void wait_async_0() {
  asm volatile("s_wait_asynccnt 0x0" ::: "memory");
}

// ---------------- init: zero output, counters, zero-page, assignment list -----
__global__ void init_kernel(float* __restrict__ y, int* __restrict__ assign_tok,
                            int* __restrict__ counts, int* __restrict__ cursors,
                            float* __restrict__ zpage) {
  int i = blockIdx.x * blockDim.x + threadIdx.x;
  if (i < S * Dm) y[i] = 0.f;
  if (i < CAP)    assign_tok[i] = -1;
  if (i < Em)     { counts[i] = 0; cursors[i] = 0; }
  if (i < 64)     zpage[i] = 0.f;
}

// ---------------- f32 -> bf16 conversion -------------------------------------
__global__ void cvt_kernel(const float* __restrict__ src,
                           unsigned short* __restrict__ dst, int n) {
  int i = blockIdx.x * blockDim.x + threadIdx.x;
  if (i < n) dst[i] = f32_to_bf16(src[i]);
}

// ---------------- router: logits, top-2, softmax, expert counts ---------------
__global__ __launch_bounds__(256) void router_kernel(
    const float* __restrict__ x, const float* __restrict__ gate_w,
    int* __restrict__ counts, int* __restrict__ top_e, float* __restrict__ top_g) {
  __shared__ float gw[Em * Dm];          // 32 KB
  for (int i = threadIdx.x; i < Em * Dm; i += 256) gw[i] = gate_w[i];
  __syncthreads();
  int s = blockIdx.x * 256 + threadIdx.x;
  if (s >= S) return;
  const float* xr = x + (size_t)s * Dm;
  float acc[Em];
  #pragma unroll
  for (int e = 0; e < Em; ++e) acc[e] = 0.f;
  for (int d = 0; d < Dm; ++d) {
    float xv = xr[d];
    #pragma unroll
    for (int e = 0; e < Em; ++e) acc[e] = __builtin_fmaf(xv, gw[e * Dm + d], acc[e]);
  }
  int e0 = 0; float v0 = acc[0];
  #pragma unroll
  for (int e = 1; e < Em; ++e) if (acc[e] > v0) { v0 = acc[e]; e0 = e; }
  int e1 = 0; float v1 = -3.4e38f;
  #pragma unroll
  for (int e = 0; e < Em; ++e) if (e != e0 && acc[e] > v1) { v1 = acc[e]; e1 = e; }
  float t  = __expf(v1 - v0);
  float g0 = 1.f / (1.f + t);
  float g1 = t * g0;
  top_e[s * 2 + 0] = e0; top_e[s * 2 + 1] = e1;
  top_g[s * 2 + 0] = g0; top_g[s * 2 + 1] = g1;
  atomicAdd(&counts[e0], 1);
  atomicAdd(&counts[e1], 1);
}

// ---------------- padded prefix offsets (tiles never straddle experts) --------
__global__ void offsets_kernel(const int* __restrict__ counts, int* __restrict__ offsets) {
  if (threadIdx.x == 0) {
    int o = 0;
    for (int e = 0; e < Em; ++e) {
      offsets[e] = o;
      o += ((counts[e] + TM - 1) / TM) * TM;
    }
    offsets[Em] = o;
  }
}

// ---------------- scatter tokens into compacted per-expert list ---------------
__global__ void scatter_kernel(const int* __restrict__ top_e, const float* __restrict__ top_g,
                               const int* __restrict__ offsets, int* __restrict__ cursors,
                               int* __restrict__ assign_tok, float* __restrict__ assign_gate) {
  int s = blockIdx.x * 256 + threadIdx.x;
  if (s >= S) return;
  #pragma unroll
  for (int j = 0; j < 2; ++j) {
    int e    = top_e[s * 2 + j];
    int slot = atomicAdd(&cursors[e], 1);
    int pos  = offsets[e] + slot;
    assign_tok[pos]  = s;
    assign_gate[pos] = top_g[s * 2 + j];
  }
}

// ---------------- GEMM1: h = relu(x @ w1[e]^T + b1[e]) -> bf16 ----------------
__global__ __launch_bounds__(256) void gemm1_kernel(
    const unsigned short* __restrict__ xb, const unsigned short* __restrict__ w1b,
    const float* __restrict__ b1, const int* __restrict__ offsets,
    const int* __restrict__ assign_tok, const float* __restrict__ zpage,
    unsigned short* __restrict__ hb) {
  __shared__ __bf16 As[2][TM * LDT];
  __shared__ __bf16 Bs[2][TM * LDT];
  const int row0  = blockIdx.x * TM;
  const int total = offsets[Em];
  if (row0 >= total) return;
  int e = 0;
  #pragma unroll
  for (int q = 1; q < Em; ++q) if (row0 >= offsets[q]) e = q;
  const int n0 = blockIdx.y * TM;
  const unsigned short* wbase = w1b + (size_t)e * Fm * Dm;

  const int tid  = threadIdx.x;
  const int wave = tid >> 5, lane = tid & 31;
  const int wm = wave & 3, wn = wave >> 2;       // 4 M-strips x 2 N-strips
  const int lh = lane >> 4, l16 = lane & 15;
  const int lr = tid >> 1,  hf  = tid & 1;       // tile load: row, 16-elem half
  const int tokr = assign_tok[row0 + lr];

  // async-copy source pointers (byte) + per-K-step stride; padding rows read a
  // zero page with stride 0
  const char* a_base = (tokr >= 0)
      ? (const char*)(xb + (size_t)tokr * Dm + hf * 16)
      : (const char*)zpage + hf * 32;
  const size_t a_step = (tokr >= 0) ? 64 : 0;
  const char* b_base = (const char*)(wbase + (size_t)(n0 + lr) * Dm + hf * 16);
  const unsigned ldsA = (unsigned)(uintptr_t)&As[0][lr * LDT + hf * 16];
  const unsigned ldsB = (unsigned)(uintptr_t)&Bs[0][lr * LDT + hf * 16];

  v8f acc[2][4] = {};
  const int nsteps = Dm / 32;
  // stage 0
  async_ld16(ldsA, a_base);
  async_ld16(ldsB, b_base);
  for (int step = 0; step < nsteps; ++step) {
    const int cur = step & 1;
    if (step + 1 < nsteps) {
      async_ld16(ldsA + (cur ^ 1) * BUFB, a_base + (size_t)(step + 1) * a_step);
      async_ld16(ldsB + (cur ^ 1) * BUFB, b_base + (size_t)(step + 1) * 64);
      wait_async_le2();
    } else {
      wait_async_0();
    }
    __syncthreads();
    ABFrag bf[4];
    #pragma unroll
    for (int sn = 0; sn < 4; ++sn) {
      const __bf16* bp = &Bs[cur][(wn * 64 + sn * 16 + l16) * LDT + lh * 16];
      bf[sn].h[0] = *(const v8bf*)(bp);
      bf[sn].h[1] = *(const v8bf*)(bp + 8);
    }
    #pragma unroll
    for (int sm = 0; sm < 2; ++sm) {
      ABFrag a;
      const __bf16* ap = &As[cur][(wm * 32 + sm * 16 + l16) * LDT];
      a.h[0] = *(const v8bf*)(ap + lh * 16);
      a.h[1] = *(const v8bf*)(ap + lh * 16 + 8);
      #pragma unroll
      for (int sn = 0; sn < 4; ++sn)
        acc[sm][sn] = __builtin_amdgcn_wmma_f32_16x16x32_bf16(
            false, a.v, false, bf[sn].v, (short)0, acc[sm][sn], false, false);
    }
    __syncthreads();
  }
  #pragma unroll
  for (int sm = 0; sm < 2; ++sm) {
    #pragma unroll
    for (int sn = 0; sn < 4; ++sn) {
      int nf = n0 + wn * 64 + sn * 16 + l16;
      float bias = b1[e * Fm + nf];
      #pragma unroll
      for (int r = 0; r < 8; ++r) {
        int row = row0 + wm * 32 + sm * 16 + lh * 8 + r;  // C: M = r + 8*h
        float hv = acc[sm][sn][r] + bias;
        hv = hv > 0.f ? hv : 0.f;
        hb[(size_t)row * Fm + nf] = f32_to_bf16(hv);
      }
    }
  }
}

// ---------------- GEMM2: y[s] += gate * (h @ w2[e]^T + b2[e]) -----------------
__global__ __launch_bounds__(256) void gemm2_kernel(
    const unsigned short* __restrict__ hb, const unsigned short* __restrict__ w2b,
    const float* __restrict__ b2, const int* __restrict__ offsets,
    const int* __restrict__ assign_tok, const float* __restrict__ assign_gate,
    float* __restrict__ y) {
  __shared__ __bf16 As[2][TM * LDT];
  __shared__ __bf16 Bs[2][TM * LDT];
  const int row0  = blockIdx.x * TM;
  const int total = offsets[Em];
  if (row0 >= total) return;
  int e = 0;
  #pragma unroll
  for (int q = 1; q < Em; ++q) if (row0 >= offsets[q]) e = q;
  const int n0 = blockIdx.y * TM;
  const unsigned short* wbase = w2b + (size_t)e * Dm * Fm;

  const int tid  = threadIdx.x;
  const int wave = tid >> 5, lane = tid & 31;
  const int wm = wave & 3, wn = wave >> 2;
  const int lh = lane >> 4, l16 = lane & 15;
  const int lr = tid >> 1,  hf  = tid & 1;

  const char* a_base = (const char*)(hb + (size_t)(row0 + lr) * Fm + hf * 16);
  const char* b_base = (const char*)(wbase + (size_t)(n0 + lr) * Fm + hf * 16);
  const unsigned ldsA = (unsigned)(uintptr_t)&As[0][lr * LDT + hf * 16];
  const unsigned ldsB = (unsigned)(uintptr_t)&Bs[0][lr * LDT + hf * 16];

  v8f acc[2][4] = {};
  const int nsteps = Fm / 32;
  async_ld16(ldsA, a_base);
  async_ld16(ldsB, b_base);
  for (int step = 0; step < nsteps; ++step) {
    const int cur = step & 1;
    if (step + 1 < nsteps) {
      async_ld16(ldsA + (cur ^ 1) * BUFB, a_base + (size_t)(step + 1) * 64);
      async_ld16(ldsB + (cur ^ 1) * BUFB, b_base + (size_t)(step + 1) * 64);
      wait_async_le2();
    } else {
      wait_async_0();
    }
    __syncthreads();
    ABFrag bf[4];
    #pragma unroll
    for (int sn = 0; sn < 4; ++sn) {
      const __bf16* bp = &Bs[cur][(wn * 64 + sn * 16 + l16) * LDT + lh * 16];
      bf[sn].h[0] = *(const v8bf*)(bp);
      bf[sn].h[1] = *(const v8bf*)(bp + 8);
    }
    #pragma unroll
    for (int sm = 0; sm < 2; ++sm) {
      ABFrag a;
      const __bf16* ap = &As[cur][(wm * 32 + sm * 16 + l16) * LDT];
      a.h[0] = *(const v8bf*)(ap + lh * 16);
      a.h[1] = *(const v8bf*)(ap + lh * 16 + 8);
      #pragma unroll
      for (int sn = 0; sn < 4; ++sn)
        acc[sm][sn] = __builtin_amdgcn_wmma_f32_16x16x32_bf16(
            false, a.v, false, bf[sn].v, (short)0, acc[sm][sn], false, false);
    }
    __syncthreads();
  }
  #pragma unroll
  for (int sm = 0; sm < 2; ++sm) {
    #pragma unroll
    for (int sn = 0; sn < 4; ++sn) {
      int nd = n0 + wn * 64 + sn * 16 + l16;
      float bias = b2[e * Dm + nd];
      #pragma unroll
      for (int r = 0; r < 8; ++r) {
        int row = row0 + wm * 32 + sm * 16 + lh * 8 + r;
        int s   = assign_tok[row];
        float g = assign_gate[row];
        float ov = (acc[sm][sn][r] + bias) * g;
        if (s >= 0) atomicAdd(&y[(size_t)s * Dm + nd], ov);
      }
    }
  }
}

// ---------------- launcher ----------------------------------------------------
extern "C" void kernel_launch(void* const* d_in, const int* in_sizes, int n_in,
                              void* d_out, int out_size, void* d_ws, size_t ws_size,
                              hipStream_t stream) {
  const float* x  = (const float*)d_in[0];
  const float* gw = (const float*)d_in[1];
  const float* w1 = (const float*)d_in[2];
  const float* b1 = (const float*)d_in[3];
  const float* w2 = (const float*)d_in[4];
  const float* b2 = (const float*)d_in[5];
  float* y = (float*)d_out;
  char*  ws = (char*)d_ws;

  size_t off = 0;
  auto alloc = [&](size_t bytes) {
    size_t o = off; off += (bytes + 255) & ~(size_t)255; return o;
  };
  int*   counts      = (int*)  (ws + alloc(Em * 4));
  int*   cursors     = (int*)  (ws + alloc(Em * 4));
  int*   offsets     = (int*)  (ws + alloc((Em + 1) * 4));
  float* zpage       = (float*)(ws + alloc(64 * 4));
  int*   top_e       = (int*)  (ws + alloc((size_t)S * 2 * 4));
  float* top_g       = (float*)(ws + alloc((size_t)S * 2 * 4));
  int*   assign_tok  = (int*)  (ws + alloc((size_t)CAP * 4));
  float* assign_gate = (float*)(ws + alloc((size_t)CAP * 4));
  unsigned short* xb  = (unsigned short*)(ws + alloc((size_t)S * Dm * 2));
  unsigned short* w1b = (unsigned short*)(ws + alloc((size_t)Em * Fm * Dm * 2));
  unsigned short* w2b = (unsigned short*)(ws + alloc((size_t)Em * Dm * Fm * 2));
  unsigned short* hb  = (unsigned short*)(ws + alloc((size_t)CAP * Fm * 2));

  init_kernel<<<(S * Dm + 255) / 256, 256, 0, stream>>>(y, assign_tok, counts,
                                                        cursors, zpage);
  cvt_kernel<<<(S * Dm + 255) / 256, 256, 0, stream>>>(x, xb, S * Dm);
  cvt_kernel<<<(Em * Fm * Dm + 255) / 256, 256, 0, stream>>>(w1, w1b, Em * Fm * Dm);
  cvt_kernel<<<(Em * Dm * Fm + 255) / 256, 256, 0, stream>>>(w2, w2b, Em * Dm * Fm);
  router_kernel<<<S / 256, 256, 0, stream>>>(x, gw, counts, top_e, top_g);
  offsets_kernel<<<1, 32, 0, stream>>>(counts, offsets);
  scatter_kernel<<<S / 256, 256, 0, stream>>>(top_e, top_g, offsets, cursors,
                                              assign_tok, assign_gate);
  gemm1_kernel<<<dim3(CAP / TM, Fm / TM), 256, 0, stream>>>(xb, w1b, b1, offsets,
                                                            assign_tok, zpage, hb);
  gemm2_kernel<<<dim3(CAP / TM, Dm / TM), 256, 0, stream>>>(hb, w2b, b2, offsets,
                                                            assign_tok, assign_gate, y);
}
